// MultiLayerEdgeGAT_25280177504677
// MI455X (gfx1250) — compile-verified, matrix operand order checked by reading, not run on hardware
//
#include <hip/hip_runtime.h>

#define N_NODES 50000
#define N_EDGES 800000
#define FDIM    128          // Fin == H*D == 128 for every layer
#define EDGE_F  16
#define NEG_SLOPE 0.2f

typedef __attribute__((ext_vector_type(16))) __bf16 v16bf;
typedef __attribute__((ext_vector_type(8)))  __bf16 v8bf;
typedef __attribute__((ext_vector_type(8)))  float  v8f;

// ---------------------------------------------------------------------------
// fp32 -> packed bf16 (one-time per layer-input)
// ---------------------------------------------------------------------------
__global__ void f32_to_bf16(const float* __restrict__ in,
                            __bf16* __restrict__ out, int n) {
  int t = blockIdx.x * blockDim.x + threadIdx.x;
  if (t < n) out[t] = (__bf16)in[t];
}

// W [K=128, C=128] row-major fp32  ->  Wt [C=128, K=128] row-major bf16
__global__ void transpose_w_bf16(const float* __restrict__ W,
                                 __bf16* __restrict__ Wt) {
  int t = blockIdx.x * blockDim.x + threadIdx.x;  // 16384
  int k = t >> 7, c = t & (FDIM - 1);
  Wt[c * FDIM + k] = (__bf16)W[t];
}

// ---------------------------------------------------------------------------
// Node GEMM: feat[N,128] = A_bf16[N,128] @ Wt_bf16^T, f32 accumulate.
// Per lane a fragment is two contiguous 16B chunks (K = 8*laneHi and 16+8*laneHi),
// so each WMMA step is 4x global_load_b128 + v_wmma_f32_16x16x32_bf16.
// grid = (3125, 2), block = 128 (4 waves; wave w -> col tile blockIdx.y*4+w)
// ---------------------------------------------------------------------------
__global__ void gemm_node_wmma(const __bf16* __restrict__ A,
                               const __bf16* __restrict__ Bt,
                               float* __restrict__ C) {
  const int lane    = threadIdx.x & 31;
  const int wave    = threadIdx.x >> 5;
  const int row0    = blockIdx.x * 16;
  const int col0    = (blockIdx.y * 4 + wave) * 16;
  const int laneLow = lane & 15;
  const int laneHi  = lane >> 4;

  const __bf16* arow = A  + (size_t)(row0 + laneLow) * FDIM + 8 * laneHi;
  const __bf16* brow = Bt + (size_t)(col0 + laneLow) * FDIM + 8 * laneHi;

  v8f acc = {};
#pragma unroll
  for (int kb = 0; kb < FDIM; kb += 32) {
    v8bf a0 = *(const v8bf*)(arow + kb);
    v8bf a1 = *(const v8bf*)(arow + kb + 16);
    v8bf b0 = *(const v8bf*)(brow + kb);
    v8bf b1 = *(const v8bf*)(brow + kb + 16);
    v16bf af, bf;
#pragma unroll
    for (int i = 0; i < 8; ++i) {
      af[i] = a0[i]; af[i + 8] = a1[i];
      bf[i] = b0[i]; bf[i + 8] = b1[i];
    }
    acc = __builtin_amdgcn_wmma_f32_16x16x32_bf16(
        /*neg_a=*/false, af, /*neg_b=*/false, bf,
        /*c_mod=*/(short)0, acc, /*reuse_a=*/false, /*reuse_b=*/false);
  }

#pragma unroll
  for (int v = 0; v < 8; ++v)   // C/D: row = v + 8*laneHi, col = laneLow
    C[(size_t)(row0 + v + 8 * laneHi) * FDIM + col0 + laneLow] = acc[v];
}

// el[n,h] = feat[n,h,:]·al[h,:], er likewise. one thread per (n,h)
__global__ void node_attn(const float* __restrict__ feat,
                          const float* __restrict__ al,
                          const float* __restrict__ ar,
                          float* __restrict__ el, float* __restrict__ er,
                          int H, int D) {
  int t = blockIdx.x * blockDim.x + threadIdx.x;
  if (t >= N_NODES * H) return;
  int n = t / H, h = t % H;
  const float* f = feat + (size_t)n * FDIM + h * D;
  float sl = 0.f, sr = 0.f;
  for (int d = 0; d < D; ++d) {
    float v = f[d];
    sl += v * al[h * D + d];
    sr += v * ar[h * D + d];
  }
  el[t] = sl; er[t] = sr;
}

// we_ae[f,h] = sum_d We[f, h*D+d] * ae[h,d]   (folds ef@We@ae -> ef@we_ae)
__global__ void weae_kernel(const float* __restrict__ We,
                            const float* __restrict__ ae,
                            float* __restrict__ weae, int H, int D) {
  int t = threadIdx.x;
  if (t >= EDGE_F * H) return;
  int f = t / H, h = t % H;
  float s = 0.f;
  for (int d = 0; d < D; ++d)
    s += We[f * FDIM + h * D + d] * ae[h * D + d];
  weae[f * H + h] = s;
}

// ee[e,h] = ef[e,:]·we_ae[:,h]
__global__ void edge_ee(const float* __restrict__ ef,
                        const float* __restrict__ weae,
                        float* __restrict__ ee, int H) {
  int e = blockIdx.x * blockDim.x + threadIdx.x;
  if (e >= N_EDGES) return;
  float x[EDGE_F];
#pragma unroll
  for (int f = 0; f < EDGE_F; ++f) x[f] = ef[(size_t)e * EDGE_F + f];
  for (int h = 0; h < H; ++h) {
    float s = 0.f;
#pragma unroll
    for (int f = 0; f < EDGE_F; ++f) s += x[f] * weae[f * H + h];
    ee[(size_t)e * H + h] = s;
  }
}

__device__ __forceinline__ unsigned flt_flip(float v) {
  unsigned u = __float_as_uint(v);
  return (u & 0x80000000u) ? ~u : (u | 0x80000000u);
}
__device__ __forceinline__ float flt_unflip(unsigned u) {
  return (u & 0x80000000u) ? __uint_as_float(u & 0x7FFFFFFFu)
                           : __uint_as_float(~u);
}

// leaky-relu logits + segment max via monotonic-uint atomicMax (global_atomic_max_u32)
__global__ void edge_logits(const float* __restrict__ el, const float* __restrict__ er,
                            const float* __restrict__ ee,
                            const int* __restrict__ src, const int* __restrict__ dst,
                            float* __restrict__ elog, unsigned* __restrict__ emaxf,
                            int H) {
  int e = blockIdx.x * blockDim.x + threadIdx.x;
  if (e >= N_EDGES) return;
  int s = src[e], d = dst[e];
  for (int h = 0; h < H; ++h) {
    float v = el[s * H + h] + er[d * H + h] + ee[(size_t)e * H + h];
    v = v > 0.f ? v : NEG_SLOPE * v;
    elog[(size_t)e * H + h] = v;
    atomicMax(&emaxf[d * H + h], flt_flip(v));
  }
}

// ex = exp(e - max), segment sum via atomicAdd (global_atomic_add_f32); in-place over elog
__global__ void edge_exp(float* __restrict__ elog,
                         const unsigned* __restrict__ emaxf,
                         float* __restrict__ esum,
                         const int* __restrict__ dst, int H) {
  int e = blockIdx.x * blockDim.x + threadIdx.x;
  if (e >= N_EDGES) return;
  int d = dst[e];
  for (int h = 0; h < H; ++h) {
    float m = flt_unflip(emaxf[d * H + h]);
    float x = __expf(elog[(size_t)e * H + h] - m);
    elog[(size_t)e * H + h] = x;
    atomicAdd(&esum[d * H + h], x);
  }
}

// one wave per edge: rst[dst, c] += feat[src, c] * a[e, c>>log2D], c = 0..127
// edge index forced wave-uniform -> src/dst/ex/esum become scalar loads
__global__ void edge_aggregate(const float* __restrict__ feat,
                               const float* __restrict__ ex,
                               const float* __restrict__ esum,
                               const int* __restrict__ src, const int* __restrict__ dst,
                               float* __restrict__ rst, int H, int log2D) {
  int lane = threadIdx.x & 31;
  int e = __builtin_amdgcn_readfirstlane(
      (int)((blockIdx.x * blockDim.x + threadIdx.x) >> 5));
  if (e >= N_EDGES) return;
  int s = src[e], d = dst[e];
#pragma unroll
  for (int i = 0; i < 4; ++i) {
    int h = (i * 32) >> log2D;                       // wave-uniform head index
    float a = ex[(size_t)e * H + h] / esum[d * H + h];
    int c = i * 32 + lane;
    atomicAdd(&rst[(size_t)d * FDIM + c], feat[(size_t)s * FDIM + c] * a);
  }
}

// bias + optional relu; writes fp32 (final layer) or packed bf16 (feeds next GEMM)
__global__ void bias_act(const float* __restrict__ rst,
                         const float* __restrict__ b,
                         float* __restrict__ outf,
                         __bf16* __restrict__ outb, int act) {
  int t = blockIdx.x * blockDim.x + threadIdx.x;
  if (t >= N_NODES * FDIM) return;
  float v = rst[t] + b[t & (FDIM - 1)];
  if (act) v = fmaxf(v, 0.f);
  if (outf) outf[t] = v;
  if (outb) outb[t] = (__bf16)v;
}

// ---------------------------------------------------------------------------
static void run_layer(const __bf16* a_bf, const float* ef,
                      const int* src, const int* dst,
                      const float* W, const float* We, const float* al,
                      const float* ar, const float* ae, const float* b,
                      int H, int D, int log2D, int act,
                      float* feat, float* rst, float* el, float* er,
                      float* ee, float* elog, unsigned* emaxf, float* esum,
                      float* weae, __bf16* wt_bf,
                      float* out_f32, __bf16* out_bf16, hipStream_t stream) {
  transpose_w_bf16<<<(FDIM * FDIM) / 256, 256, 0, stream>>>(W, wt_bf);
  gemm_node_wmma<<<dim3(N_NODES / 16, 2), 128, 0, stream>>>(a_bf, wt_bf, feat);

  int nh = N_NODES * H;
  node_attn<<<(nh + 255) / 256, 256, 0, stream>>>(feat, al, ar, el, er, H, D);
  weae_kernel<<<1, 64, 0, stream>>>(We, ae, weae, H, D);
  edge_ee<<<(N_EDGES + 255) / 256, 256, 0, stream>>>(ef, weae, ee, H);

  hipMemsetAsync(emaxf, 0, sizeof(unsigned) * nh, stream);  // 0 == flipped minimum
  hipMemsetAsync(esum, 0, sizeof(float) * nh, stream);
  hipMemsetAsync(rst, 0, sizeof(float) * (size_t)N_NODES * FDIM, stream);

  edge_logits<<<(N_EDGES + 255) / 256, 256, 0, stream>>>(el, er, ee, src, dst,
                                                         elog, emaxf, H);
  edge_exp<<<(N_EDGES + 255) / 256, 256, 0, stream>>>(elog, emaxf, esum, dst, H);
  edge_aggregate<<<(N_EDGES + 7) / 8, 256, 0, stream>>>(feat, elog, esum, src, dst,
                                                        rst, H, log2D);
  bias_act<<<(N_NODES * FDIM + 255) / 256, 256, 0, stream>>>(rst, b, out_f32,
                                                             out_bf16, act);
}

extern "C" void kernel_launch(void* const* d_in, const int* in_sizes, int n_in,
                              void* d_out, int out_size, void* d_ws, size_t ws_size,
                              hipStream_t stream) {
  (void)in_sizes; (void)n_in; (void)out_size; (void)ws_size;
  const float* h   = (const float*)d_in[0];
  const float* ef  = (const float*)d_in[1];
  const int*   src = (const int*)d_in[2];
  const int*   dst = (const int*)d_in[3];

  const float *W[3], *We[3], *al[3], *ar[3], *ae[3], *b[3];
  for (int i = 0; i < 3; ++i) {
    W[i]  = (const float*)d_in[4 + 6 * i];
    We[i] = (const float*)d_in[5 + 6 * i];
    al[i] = (const float*)d_in[6 + 6 * i];
    ar[i] = (const float*)d_in[7 + 6 * i];
    ae[i] = (const float*)d_in[8 + 6 * i];
    b[i]  = (const float*)d_in[9 + 6 * i];
  }

  // workspace carve-up (fp32 first, then bf16; all offsets stay 16B-aligned)
  float*    feat  = (float*)d_ws;                               // N*128 f32
  float*    rst   = feat + (size_t)N_NODES * FDIM;              // N*128 f32
  float*    el    = rst + (size_t)N_NODES * FDIM;               // N*4
  float*    er    = el + (size_t)N_NODES * 4;                   // N*4
  float*    ee    = er + (size_t)N_NODES * 4;                   // E*4
  float*    elog  = ee + (size_t)N_EDGES * 4;                   // E*4
  unsigned* emaxf = (unsigned*)(elog + (size_t)N_EDGES * 4);    // N*4
  float*    esum  = (float*)(emaxf + (size_t)N_NODES * 4);      // N*4
  float*    weae  = esum + (size_t)N_NODES * 4;                 // 64
  __bf16*   a_bf  = (__bf16*)(weae + 64);                       // N*128 bf16
  __bf16*   wt_bf = a_bf + (size_t)N_NODES * FDIM;              // 128*128 bf16

  // pack external input once
  f32_to_bf16<<<(N_NODES * FDIM + 255) / 256, 256, 0, stream>>>(
      h, a_bf, N_NODES * FDIM);

  // layer 0: H=4, D=32, relu            a_bf -> a_bf (bf16)
  run_layer(a_bf, ef, src, dst, W[0], We[0], al[0], ar[0], ae[0], b[0],
            4, 32, 5, 1, feat, rst, el, er, ee, elog, emaxf, esum, weae,
            wt_bf, nullptr, a_bf, stream);
  // layer 1: H=4, D=32, relu            a_bf -> a_bf (bf16)
  run_layer(a_bf, ef, src, dst, W[1], We[1], al[1], ar[1], ae[1], b[1],
            4, 32, 5, 1, feat, rst, el, er, ee, elog, emaxf, esum, weae,
            wt_bf, nullptr, a_bf, stream);
  // layer 2: H=1, D=128, no act         a_bf -> d_out (fp32)
  run_layer(a_bf, ef, src, dst, W[2], We[2], al[2], ar[2], ae[2], b[2],
            1, 128, 7, 0, feat, rst, el, er, ee, elog, emaxf, esum, weae,
            wt_bf, (float*)d_out, nullptr, stream);
}
